// MambaBlock_60172491817002
// MI455X (gfx1250) — matrix operand order repaired and checked by reading, not compiled, not run on hardware
//
#include <hip/hip_runtime.h>
#include <hip/hip_bf16.h>
#include <math.h>

typedef __attribute__((ext_vector_type(16))) _Float16 v16h;
typedef __attribute__((ext_vector_type(8)))  float    v8f;

#define BATCH 2
#define EE    192
#define LLEN  8000
#define BL    (BATCH * LLEN)   // 16000 rows, divisible by 16

__device__ __forceinline__ float sigmoidf_(float x) { return 1.f / (1.f + __expf(-x)); }

__device__ __forceinline__ float act_apply(float x, int act) {
  if (act == 1) return (x > 20.f) ? x : log1pf(__expf(x));                  // softplus
  if (act == 2) return 0.5f * x * (1.f + erff(x * 0.70710678118654752f));    // exact GELU
  return x;
}

__device__ __forceinline__ void cvt4(v16h& d, int base, float4 f) {
  d[base + 0] = (_Float16)f.x; d[base + 1] = (_Float16)f.y;
  d[base + 2] = (_Float16)f.z; d[base + 3] = (_Float16)f.w;
}

// ---------------------------------------------------------------------------
// Guard-free WMMA GEMM: C[M x N] = act(A[M x K] * W^T + bias).
// W is (N x K) row-major, K % 32 == 0 (compile-time), N % 16 == 0.
// One wave per 16x16 C tile; 4 waves/block cover 4 N-tiles.
// A fragment per lane = two contiguous 8-float runs; B = the same two 8-float runs
// (per the CDNA5 16-bit WMMA VGPR layout tables) -> b128 vector loads.
// ---------------------------------------------------------------------------
template <int K>
__global__ void k_gemm(const float* __restrict__ A, int lda,
                       const float* __restrict__ W,
                       const float* __restrict__ bias,
                       float* __restrict__ C, int ldc,
                       int Nt, int act, int store_cf, int Nout) {
  const int lane = threadIdx.x & 31;
  const int wave = threadIdx.x >> 5;
  const int half = lane >> 4;
  const int r    = lane & 15;
  const int ntile = blockIdx.y * 4 + wave;
  if (ntile >= Nt) return;                    // wave-uniform
  const int mbase = blockIdx.x << 4;
  const int nbase = ntile << 4;
  const float* Arow = A + (size_t)(mbase + r) * lda;
  const float* Wrow = W + (size_t)(nbase + r) * K;

  v8f c = {0.f, 0.f, 0.f, 0.f, 0.f, 0.f, 0.f, 0.f};
#pragma unroll
  for (int ks = 0; ks < K / 32; ++ks) {
    const int kb = ks * 32;
    const float4 a0 = *(const float4*)(Arow + kb + half * 8);
    const float4 a1 = *(const float4*)(Arow + kb + half * 8 + 4);
    const float4 a2 = *(const float4*)(Arow + kb + 16 + half * 8);
    const float4 a3 = *(const float4*)(Arow + kb + 16 + half * 8 + 4);
    const float4 b0 = *(const float4*)(Wrow + kb + half * 8);
    const float4 b1 = *(const float4*)(Wrow + kb + half * 8 + 4);
    const float4 b2 = *(const float4*)(Wrow + kb + 16 + half * 8);
    const float4 b3 = *(const float4*)(Wrow + kb + 16 + half * 8 + 4);
    v16h a, b;
    cvt4(a, 0, a0); cvt4(a, 4, a1); cvt4(a, 8, a2); cvt4(a, 12, a3);
    cvt4(b, 0, b0); cvt4(b, 4, b1); cvt4(b, 8, b2); cvt4(b, 12, b3);
    c = __builtin_amdgcn_wmma_f32_16x16x32_f16(false, a, false, b, (short)0, c,
                                               false, false);
  }
  const int col = nbase + r;
  const float bv = bias ? bias[col] : 0.f;
#pragma unroll
  for (int rr = 0; rr < 8; ++rr) {
    const int row = mbase + rr + (half << 3);
    const float val = act_apply(c[rr] + bv, act);
    if (store_cf) {          // channel-first (B, N, L) final output
      const int bb = row / LLEN, ll = row % LLEN;
      C[((size_t)bb * Nout + col) * LLEN + ll] = val;
    } else {
      C[(size_t)row * ldc + col] = val;
    }
  }
}

// ---------------------------------------------------------------------------
// Downsample conv as implicit GEMM: rows = 16000 voxels, K = 384 gathered
// patch elements, N = 96 output channels. A gathered per-lane, B vectorized.
// ---------------------------------------------------------------------------
__global__ void k_conv_gemm(const float* __restrict__ X,   // (2,48,40,40,40)
                            const float* __restrict__ Wt,  // (96,384)
                            const float* __restrict__ bias,
                            float* __restrict__ C) {       // (BL,96)
  const int lane = threadIdx.x & 31;
  const int wave = threadIdx.x >> 5;
  const int half = lane >> 4;
  const int r    = lane & 15;
  const int ntile = blockIdx.y * 4 + wave;
  if (ntile >= 6) return;
  const int mbase = blockIdx.x << 4;
  const int nbase = ntile << 4;
  const int m   = mbase + r;
  const int col = nbase + r;
  const int bb = m / LLEN;
  const int l  = m % LLEN;
  const int d  = l / 400;
  const int h  = (l / 20) % 20;
  const int w  = l % 20;
  const float* Xb = X + (size_t)bb * 48 * 64000;            // 40^3
  const int base  = ((2 * d) * 40 + 2 * h) * 40 + 2 * w;
  const float* Wrow = Wt + (size_t)col * 384;

  v8f c = {0.f, 0.f, 0.f, 0.f, 0.f, 0.f, 0.f, 0.f};
  for (int ks = 0; ks < 12; ++ks) {
    const int kb = ks * 32;
    v16h a, b;
#pragma unroll
    for (int i = 0; i < 16; ++i) {
      const int k   = kb + half * 16 + i;
      const int ch  = k >> 3;
      const int rem = k & 7;
      const int kd = rem >> 2, kh = (rem >> 1) & 1, kw = rem & 1;
      a[i] = (_Float16)Xb[(size_t)ch * 64000 + base + (kd * 40 + kh) * 40 + kw];
    }
    const float4 b0 = *(const float4*)(Wrow + kb + half * 16);
    const float4 b1 = *(const float4*)(Wrow + kb + half * 16 + 4);
    const float4 b2 = *(const float4*)(Wrow + kb + half * 16 + 8);
    const float4 b3 = *(const float4*)(Wrow + kb + half * 16 + 12);
    cvt4(b, 0, b0); cvt4(b, 4, b1); cvt4(b, 8, b2); cvt4(b, 12, b3);
    c = __builtin_amdgcn_wmma_f32_16x16x32_f16(false, a, false, b, (short)0, c,
                                               false, false);
  }
  const float bv = bias[col];
#pragma unroll
  for (int rr = 0; rr < 8; ++rr) {
    const int row = mbase + rr + (half << 3);
    C[(size_t)row * 96 + col] = c[rr] + bv;
  }
}

// ---------------------------------------------------------------------------
// Pad weight (rows_src x K) -> (rows_dst x K) with zero rows (for N=38->48).
// ---------------------------------------------------------------------------
__global__ void k_pad_w(const float* __restrict__ src, float* __restrict__ dst,
                        int rows_src, int rows_dst, int K) {
  const int t = blockIdx.x * blockDim.x + threadIdx.x;
  if (t >= rows_dst * K) return;
  const int rr = t / K;
  dst[t] = (rr < rows_src) ? src[t] : 0.f;
}

// ---------------------------------------------------------------------------
// Row layer-norm over C=96 features.
// ---------------------------------------------------------------------------
__global__ void k_ln_row(const float* __restrict__ in, const float* __restrict__ w,
                         const float* __restrict__ b, float* __restrict__ out,
                         int M, int Cdim, float eps) {
  const int row = blockIdx.x * blockDim.x + threadIdx.x;
  if (row >= M) return;
  const float* p = in + (size_t)row * Cdim;
  float s = 0.f;
  for (int i = 0; i < Cdim; ++i) s += p[i];
  const float mean = s / Cdim;
  float v = 0.f;
  for (int i = 0; i < Cdim; ++i) { const float d = p[i] - mean; v += d * d; }
  const float inv = rsqrtf(v / Cdim + eps);
  float* o = out + (size_t)row * Cdim;
  for (int i = 0; i < Cdim; ++i) o[i] = (p[i] - mean) * inv * w[i] + b[i];
}

// ---------------------------------------------------------------------------
// Causal depthwise conv1d (K=4) + bias + SiLU, optional sequence flip.
// ---------------------------------------------------------------------------
__global__ void k_conv1d_silu(const float* __restrict__ xz,  // (BL,384), xi = cols 0..191
                              const float* __restrict__ cw,  // (E,4)
                              const float* __restrict__ cb,  // (E)
                              float* __restrict__ xc, int dir) {
  const int t = blockIdx.x * blockDim.x + threadIdx.x;
  if (t >= BL * EE) return;
  const int e  = t % EE;
  const int bl = t / EE;
  const int b  = bl / LLEN;
  const int l  = bl % LLEN;
  float acc = cb[e];
#pragma unroll
  for (int j = 0; j < 4; ++j) {
    const int ls = l - 3 + j;
    if (ls >= 0) {
      const int phys = dir ? (LLEN - 1 - ls) : ls;
      acc += cw[e * 4 + j] * xz[(size_t)(b * LLEN + phys) * 384 + e];
    }
  }
  xc[(size_t)bl * EE + e] = acc * sigmoidf_(acc);
}

// ---------------------------------------------------------------------------
// dt-projection: dt = softplus(dbl[:, :6] @ dtw^T + dtb). K=6 -> VALU kernel.
// ---------------------------------------------------------------------------
__global__ void k_dtproj(const float* __restrict__ dbl,  // (BL,48)
                         const float* __restrict__ dtw,  // (192,6)
                         const float* __restrict__ dtb,  // (192)
                         float* __restrict__ dt) {       // (BL,192)
  const int t = blockIdx.x * blockDim.x + threadIdx.x;
  if (t >= BL * EE) return;
  const int e  = t % EE;
  const int bl = t / EE;
  float acc = dtb[e];
#pragma unroll
  for (int rr = 0; rr < 6; ++rr) acc += dbl[(size_t)bl * 48 + rr] * dtw[e * 6 + rr];
  dt[(size_t)bl * EE + e] = (acc > 20.f) ? acc : log1pf(__expf(acc));
}

// ---------------------------------------------------------------------------
// Selective scan: one lane per (b, e, n); 16 state lanes per (b,e) group
// reduced via xor-shuffles inside the wave32 half. Sequential over L=8000.
// ---------------------------------------------------------------------------
__global__ void k_scan(const float* __restrict__ xc,   // (BL,192)
                       const float* __restrict__ dt,   // (BL,192)
                       const float* __restrict__ dbl,  // (BL,48): [6:22]=B, [22:38]=C
                       const float* __restrict__ a_log,// (192,16)
                       const float* __restrict__ dp,   // (192)
                       float* __restrict__ y) {        // (BL,192)
  const int t = blockIdx.x * blockDim.x + threadIdx.x;
  if (t >= BATCH * EE * 16) return;
  const int n  = t & 15;
  const int ge = t >> 4;
  const int e  = ge % EE;
  const int b  = ge / EE;
  const float An = -__expf(a_log[e * 16 + n]);
  const float Dv = dp[e];
  float hst = 0.f;
  for (int l = 0; l < LLEN; ++l) {
    const size_t ro = (size_t)b * LLEN + l;
    const float dtv = dt[ro * EE + e];
    const float xv  = xc[ro * EE + e];
    const float Bv  = dbl[ro * 48 + 6 + n];
    const float Cv  = dbl[ro * 48 + 22 + n];
    const float dA  = __expf(dtv * An);
    hst = dA * hst + dtv * Bv * xv;
    float p = hst * Cv;
    p += __shfl_xor(p, 1);
    p += __shfl_xor(p, 2);
    p += __shfl_xor(p, 4);
    p += __shfl_xor(p, 8);
    if (n == 0) y[ro * EE + e] = p + Dv * xv;
  }
}

// ---------------------------------------------------------------------------
// sum = (yf[l] + yb[L-1-l]) * silu(z[l]); written in place into y0.
// ---------------------------------------------------------------------------
__global__ void k_combine(float* __restrict__ y0, const float* __restrict__ y1,
                          const float* __restrict__ xz) {
  const int t = blockIdx.x * blockDim.x + threadIdx.x;
  if (t >= BL * EE) return;
  const int e  = t % EE;
  const int bl = t / EE;
  const int b  = bl / LLEN;
  const int l  = bl % LLEN;
  const float zf = xz[(size_t)bl * 384 + EE + e];
  const float s  = zf * sigmoidf_(zf);
  const float vf = y0[(size_t)bl * EE + e];
  const float vb = y1[((size_t)b * LLEN + (LLEN - 1 - l)) * EE + e];
  y0[(size_t)bl * EE + e] = (vf + vb) * s;
}

extern "C" void kernel_launch(void* const* d_in, const int* in_sizes, int n_in,
                              void* d_out, int out_size, void* d_ws, size_t ws_size,
                              hipStream_t stream) {
  (void)in_sizes; (void)n_in; (void)out_size; (void)ws_size;
  const float* x      = (const float*)d_in[0];
  const float* ds_w   = (const float*)d_in[1];
  const float* ds_b   = (const float*)d_in[2];
  const float* ds_lnw = (const float*)d_in[3];
  const float* ds_lnb = (const float*)d_in[4];
  const float* ln_w   = (const float*)d_in[5];
  const float* ln_b   = (const float*)d_in[6];
  const float* in_w   = (const float*)d_in[7];
  const float* conv_w = (const float*)d_in[8];
  const float* conv_b = (const float*)d_in[9];
  const float* xp_w   = (const float*)d_in[10];
  const float* dtp_w  = (const float*)d_in[11];
  const float* dtp_b  = (const float*)d_in[12];
  const float* A_log  = (const float*)d_in[13];
  const float* Dp     = (const float*)d_in[14];
  const float* out_w  = (const float*)d_in[15];
  const float* f_lnw  = (const float*)d_in[16];
  const float* f_lnb  = (const float*)d_in[17];
  const float* fc1_w  = (const float*)d_in[18];
  const float* fc1_b  = (const float*)d_in[19];
  const float* fc2_w  = (const float*)d_in[20];
  const float* fc2_b  = (const float*)d_in[21];

  float* ws  = (float*)d_ws;
  float* xf  = ws;                           // (BL,96)
  float* xn  = xf  + (size_t)BL * 96;        // (BL,96)
  float* xz  = xn  + (size_t)BL * 96;        // (BL,384)  also MLP hidden
  float* xc0 = xz  + (size_t)BL * 384;       // (BL,192)
  float* xc1 = xc0 + (size_t)BL * 192;
  float* db0 = xc1 + (size_t)BL * 192;       // (BL,48), cols 38..47 zero
  float* db1 = db0 + (size_t)BL * 48;
  float* dt0 = db1 + (size_t)BL * 48;        // (BL,192)
  float* dt1 = dt0 + (size_t)BL * 192;
  float* y0  = dt1 + (size_t)BL * 192;       // (BL,192)
  float* y1  = y0  + (size_t)BL * 192;
  float* wp0 = y1  + (size_t)BL * 192;       // padded xp_w: (48,192)
  float* wp1 = wp0 + 48 * 192;

  const dim3 blk(128);                       // 4 waves: 4 N-tiles per block
  const int lng = (BL + 255) / 256;
  const int ct  = (BL * EE + 255) / 256;
  const int st  = (BATCH * EE * 16 + 255) / 256;
  const int pt  = (48 * 192 + 255) / 256;

  // Stage 1: downsample conv (implicit WMMA GEMM, K=384) + bias, channel LN
  k_conv_gemm<<<dim3(BL / 16, 2), blk, 0, stream>>>(x, ds_w, ds_b, xn);
  k_ln_row<<<lng, 256, 0, stream>>>(xn, ds_lnw, ds_lnb, xf, BL, 96, 1e-6f);

  for (int i = 0; i < 2; ++i) {
    // LN -> in-proj (K=96, N=384 -> 24 tiles)
    k_ln_row<<<lng, 256, 0, stream>>>(xf, ln_w + i * 96, ln_b + i * 96, xn, BL, 96, 1e-5f);
    k_gemm<96><<<dim3(BL / 16, 6), blk, 0, stream>>>(xn, 96, in_w + (size_t)i * 384 * 96,
                                                     nullptr, xz, 384, 24, 0, 0, 0);
    // bidirectional causal conv1d + SiLU
    k_conv1d_silu<<<ct, 256, 0, stream>>>(xz, conv_w + (i * 2 + 0) * EE * 4,
                                          conv_b + (i * 2 + 0) * EE, xc0, 0);
    k_conv1d_silu<<<ct, 256, 0, stream>>>(xz, conv_w + (i * 2 + 1) * EE * 4,
                                          conv_b + (i * 2 + 1) * EE, xc1, 1);
    // pad x-proj weights (38x192 -> 48x192, zero rows), then x-proj (K=192, N=48)
    k_pad_w<<<pt, 256, 0, stream>>>(xp_w + (size_t)(i * 2 + 0) * 38 * 192, wp0, 38, 48, 192);
    k_pad_w<<<pt, 256, 0, stream>>>(xp_w + (size_t)(i * 2 + 1) * 38 * 192, wp1, 38, 48, 192);
    k_gemm<192><<<dim3(BL / 16, 1), blk, 0, stream>>>(xc0, 192, wp0, nullptr, db0, 48, 3, 0, 0, 0);
    k_gemm<192><<<dim3(BL / 16, 1), blk, 0, stream>>>(xc1, 192, wp1, nullptr, db1, 48, 3, 0, 0, 0);
    // dt-proj + softplus (K=6: VALU kernel)
    k_dtproj<<<ct, 256, 0, stream>>>(db0, dtp_w + (size_t)(i * 2 + 0) * 192 * 6,
                                     dtp_b + (i * 2 + 0) * 192, dt0);
    k_dtproj<<<ct, 256, 0, stream>>>(db1, dtp_w + (size_t)(i * 2 + 1) * 192 * 6,
                                     dtp_b + (i * 2 + 1) * 192, dt1);
    // selective scans (both directions)
    k_scan<<<st, 256, 0, stream>>>(xc0, dt0, db0, A_log + (size_t)(i * 2 + 0) * EE * 16,
                                   Dp + (i * 2 + 0) * EE, y0);
    k_scan<<<st, 256, 0, stream>>>(xc1, dt1, db1, A_log + (size_t)(i * 2 + 1) * EE * 16,
                                   Dp + (i * 2 + 1) * EE, y1);
    // gated combine (in place into y0) then out-proj (K=192, N=96 -> 6 tiles)
    k_combine<<<ct, 256, 0, stream>>>(y0, y1, xz);
    k_gemm<192><<<dim3(BL / 16, 2), blk, 0, stream>>>(y0, 192, out_w + (size_t)i * 96 * 192,
                                                      nullptr, xf, 96, 6, 0, 0, 0);
  }

  // Final channel LN -> fc1+GELU (K=96, N=384) -> fc2 (K=384, N=96, channel-first out)
  k_ln_row<<<lng, 256, 0, stream>>>(xf, f_lnw, f_lnb, xn, BL, 96, 1e-6f);
  k_gemm<96><<<dim3(BL / 16, 6), blk, 0, stream>>>(xn, 96, fc1_w, fc1_b, xz, 384, 24, 2, 0, 0);
  k_gemm<384><<<dim3(BL / 16, 2), blk, 0, stream>>>(xz, 384, fc2_w, fc2_b, (float*)d_out,
                                                    96, 6, 0, 1, 96);
}
